// MultiHeadedEdgeAttention_62414464745780
// MI455X (gfx1250) — matrix operand order repaired and checked by reading, not compiled, not run on hardware
//
#include <hip/hip_runtime.h>
#include <hip/hip_bf16.h>
#include <stdint.h>

#define B_TOTAL 262144
#define MT      64          // rows of B per workgroup
#define TPB     256         // 8 waves (wave32)

typedef __attribute__((ext_vector_type(16))) __bf16 v16bf;
typedef __attribute__((ext_vector_type(8)))  float  v8f;

union Frag16 { v16bf v; unsigned short u[16]; uint4 q[2]; };

// LDS layout (bytes) -- total exactly 160KB
#define XT_OFF    0            // 64*256 bf16  input tile          = 32768
#define WB_OFF    32768        // 256*32 f32   weight chunk (DMA)  = 32768
#define WBH_OFF   65536        // 2x 256*32 bf16 weight chunk      = 32768
#define OQ_OFF    98304        // 64*256 bf16  Q (later V)         = 32768
#define OK_OFF    131072       // 64*256 bf16  K                   = 32768
#define AT_OFF    OK_OFF       // 64*16 f32 attn ALIASES K region (K dead by then)
#define SMEM_BYTES 163840

// ---- bf16 helpers ----------------------------------------------------------
#if __has_builtin(__builtin_amdgcn_cvt_pk_bf16_f32)
typedef __attribute__((ext_vector_type(2))) __bf16 v2bf;
__device__ __forceinline__ unsigned pack2bf(float a, float b) {
    v2bf r = __builtin_amdgcn_cvt_pk_bf16_f32(a, b);   // v_cvt_pk_bf16_f32 (1 VALU)
    return __builtin_bit_cast(unsigned, r);
}
#elif __has_builtin(__builtin_amdgcn_perm)
__device__ __forceinline__ unsigned pack2bf(float a, float b) {
    // {hi16(b), hi16(a)} in one v_perm_b32: sel bytes (7,6,3,2); 0-3 pick S1, 4-7 pick S0
    return __builtin_amdgcn_perm(__builtin_bit_cast(unsigned, b),
                                 __builtin_bit_cast(unsigned, a), 0x07060302u);
}
#else
__device__ __forceinline__ unsigned pack2bf(float a, float b) {
    unsigned ua = __builtin_bit_cast(unsigned, a);
    unsigned ub = __builtin_bit_cast(unsigned, b);
    return (ua >> 16) | (ub & 0xffff0000u);            // truncate
}
#endif
__device__ __forceinline__ unsigned short f2bf1(float f) {
    unsigned u = __builtin_bit_cast(unsigned, f);
    return (unsigned short)((u + 0x8000u) >> 16);      // 2 VALU, round-to-nearest
}
// unpack low/high bf16 of a u32 directly (1 VALU each)
__device__ __forceinline__ float bflo(unsigned p) {
    return __builtin_bit_cast(float, p << 16);
}
__device__ __forceinline__ float bfhi(unsigned p) {
    return __builtin_bit_cast(float, p & 0xffff0000u);
}

// ---- CDNA5 async DMA global -> LDS (ASYNCcnt) ------------------------------
__device__ __forceinline__ void async_copy_b128(unsigned lds_off, const void* gaddr) {
    asm volatile("global_load_async_to_lds_b128 %0, %1, off"
                 :: "v"(lds_off),
                    "v"((unsigned long long)(uintptr_t)gaddr)
                 : "memory");
}
__device__ __forceinline__ void wait_async0() {
    asm volatile("s_wait_asynccnt 0x0" ::: "memory");
}

// OUT = bf16( X(64x256) @ W^T + b ), written to LDS at out_off.
// Weight streamed in 8 K-chunks of 32 cols; DMA of chunk k+1 overlaps WMMAs of chunk k.
__device__ __forceinline__ void proj_gemm(const float* __restrict__ Xg,
                                          const float* __restrict__ Wg,
                                          const float* __restrict__ bg,
                                          unsigned char* sm, int out_off) {
    const int tid  = threadIdx.x;
    const int lane = tid & 31;
    const int w    = tid >> 5;
    const int rb   = w & 3;        // 16-row block 0..3
    const int ch   = w >> 2;       // 128-col half 0..1
    const int nlo  = lane & 15;
    const int hsel = lane >> 4;    // lane half select

    unsigned short* XT  = (unsigned short*)(sm + XT_OFF);
    float*          WB  = (float*)(sm + WB_OFF);
    unsigned short* OUT = (unsigned short*)(sm + out_off);

    const unsigned wb_ldsbase = (unsigned)(WB_OFF + tid * 128);

    // ---- issue DMA for chunk 0 first (overlaps with input-tile staging)
    {
        const float* gsrc = Wg + tid * 256;    // thread t owns W row t, 32 floats
#pragma unroll
        for (int i = 0; i < 8; ++i)
            async_copy_b128(wb_ldsbase + i * 16, gsrc + i * 4);
    }

    // ---- stage 64x256 fp32 input tile into LDS as bf16 (coalesced b128 loads)
    {
        const float4* Xg4 = (const float4*)Xg;
        unsigned* XTu = (unsigned*)XT;
#pragma unroll
        for (int i = 0; i < 16; ++i) {
            int vi = i * TPB + tid;            // float4 index 0..4095
            float4 f = Xg4[vi];
            XTu[vi * 2 + 0] = pack2bf(f.x, f.y);
            XTu[vi * 2 + 1] = pack2bf(f.z, f.w);
        }
    }
    wait_async0();
    __syncthreads();

    v8f acc[8];
#pragma unroll
    for (int nb = 0; nb < 8; ++nb) { v8f z = {}; acc[nb] = z; }

    const int am  = rb * 16 + nlo;     // A-matrix M (lane layout, ISA 7.12.2)
    const int alo = hsel * 8;          // A-matrix K base per lane-half

    for (int kc = 0; kc < 8; ++kc) {
        unsigned short* WBH = (unsigned short*)(sm + WBH_OFF + ((kc & 1) << 14));

        // ---- convert staged fp32 chunk -> bf16 double buffer (own 32 floats)
        {
            const float* src = WB + tid * 32;
            unsigned* dst = (unsigned*)WBH + tid * 16;
#pragma unroll
            for (int j = 0; j < 16; ++j)
                dst[j] = pack2bf(src[2 * j], src[2 * j + 1]);
        }
        __syncthreads();                        // WB free; WBH[kc&1] ready

        // ---- issue DMA for chunk kc+1; streams in underneath the WMMAs below
        if (kc < 7) {
            const float* gsrc = Wg + tid * 256 + (kc + 1) * 32;
#pragma unroll
            for (int i = 0; i < 8; ++i)
                async_copy_b128(wb_ldsbase + i * 16, gsrc + i * 4);
        }

        // ---- A fragment: rows am, K = kc*32 + {alo..alo+7, alo+16..alo+23}
        Frag16 a;
        {
            const uint4* p = (const uint4*)&XT[am * 256 + kc * 32 + alo];
            a.q[0] = p[0];
            a.q[1] = p[2];   // +16 bf16 elements
        }
#pragma unroll
        for (int nb = 0; nb < 8; ++nb) {
            // B fragment: B[k][n] = W[n][k]; lane holds col n=nlo, K = {hsel*8..+7, 16+hsel*8..+7}
            Frag16 b;
            const uint4* p =
                (const uint4*)&WBH[(((ch * 8 + nb) * 16 + nlo) * 32) + hsel * 8];
            b.q[0] = p[0];
            b.q[1] = p[2];
            acc[nb] = __builtin_amdgcn_wmma_f32_16x16x32_bf16(
                false, a.v, false, b.v, (short)0, acc[nb], false, false);
        }
        if (kc < 7) wait_async0();
        __syncthreads();                        // next chunk resident in WB
    }

    // ---- epilogue: + bias, store bf16 tile to OUT (C/D layout ISA 7.12.2)
#pragma unroll
    for (int nb = 0; nb < 8; ++nb) {
        int n = (ch * 8 + nb) * 16 + nlo;
        float bn = bg[n];
#pragma unroll
        for (int j = 0; j < 8; ++j) {
            int m = rb * 16 + j + 8 * hsel;
            OUT[m * 256 + n] = f2bf1(acc[nb][j] + bn);
        }
    }
    __syncthreads();
}

__global__ __launch_bounds__(TPB, 1)
void mhea_fused_kernel(const float* __restrict__ q, const float* __restrict__ k,
                       const float* __restrict__ v, const float* __restrict__ attn_bias,
                       const float* __restrict__ Wq, const float* __restrict__ bq,
                       const float* __restrict__ Wk, const float* __restrict__ bk,
                       const float* __restrict__ Wv, const float* __restrict__ bv,
                       float* __restrict__ out) {
    __shared__ __align__(16) unsigned char sm[SMEM_BYTES];
    const int tid = threadIdx.x;
    const long long b0 = (long long)blockIdx.x * MT;

    proj_gemm(q + b0 * 256, Wq, bq, sm, OQ_OFF);   // Q tile
    proj_gemm(k + b0 * 256, Wk, bk, sm, OK_OFF);   // K tile

    // ---- prob[h][g] = sum_d Q[r][4d+h]*K[r][4d+g] + bias; softmax over h (axis 1)
    {
        unsigned short* Q = (unsigned short*)(sm + OQ_OFF);
        unsigned short* K = (unsigned short*)(sm + OK_OFF);
        int r = tid >> 2, g = tid & 3;
        const uint2* Qr = (const uint2*)&Q[r * 256];
        const unsigned* Kr = (const unsigned*)&K[r * 256];
        float s0 = 0.f, s1 = 0.f, s2 = 0.f, s3 = 0.f;
#pragma unroll 8
        for (int d = 0; d < 64; ++d) {
            uint2 qp = Qr[d];
            unsigned kp = Kr[d * 2 + (g >> 1)];
            float kv = (g & 1) ? bfhi(kp) : bflo(kp);
            s0 += bflo(qp.x) * kv;
            s1 += bfhi(qp.x) * kv;
            s2 += bflo(qp.y) * kv;
            s3 += bfhi(qp.y) * kv;
        }
        const float* bias = attn_bias + (b0 + r) * 16;
        s0 += bias[0 * 4 + g]; s1 += bias[1 * 4 + g];
        s2 += bias[2 * 4 + g]; s3 += bias[3 * 4 + g];

        __syncthreads();   // K reads done; ATT below aliases the K region

        // second output: prob [B,4,4] (pre-softmax, with bias)
        float* probOut = out + (long long)B_TOTAL * 256 + (b0 + r) * 16 + g;
        probOut[0] = s0; probOut[4] = s1; probOut[8] = s2; probOut[12] = s3;

        // softmax over h for fixed g
        float* ATT = (float*)(sm + AT_OFF);
        float m = fmaxf(fmaxf(s0, s1), fmaxf(s2, s3));
        float e0 = __expf(s0 - m), e1 = __expf(s1 - m);
        float e2 = __expf(s2 - m), e3 = __expf(s3 - m);
        float inv = 1.f / (e0 + e1 + e2 + e3);
        ATT[r * 16 + 0 * 4 + g] = e0 * inv;
        ATT[r * 16 + 1 * 4 + g] = e1 * inv;
        ATT[r * 16 + 2 * 4 + g] = e2 * inv;
        ATT[r * 16 + 3 * 4 + g] = e3 * inv;
    }
    __syncthreads();

    proj_gemm(v + b0 * 256, Wv, bv, sm, OQ_OFF);   // V tile (reuse Q region)

    // ---- x[r][h][d] = sum_g attn[r][h][g] * V[r][4d+g] ; out[b, h*64+d]
    {
        unsigned short* V = (unsigned short*)(sm + OQ_OFF);
        float* ATT = (float*)(sm + AT_OFF);
        int r = tid >> 2, h = tid & 3;
        float a0 = ATT[r * 16 + h * 4 + 0];
        float a1 = ATT[r * 16 + h * 4 + 1];
        float a2 = ATT[r * 16 + h * 4 + 2];
        float a3 = ATT[r * 16 + h * 4 + 3];
        const uint2* Vr = (const uint2*)&V[r * 256];
        float4* xo = (float4*)(out + (b0 + r) * 256 + h * 64);
#pragma unroll
        for (int d4 = 0; d4 < 16; ++d4) {
            float tmp[4];
#pragma unroll
            for (int j = 0; j < 4; ++j) {
                uint2 vp = Vr[d4 * 4 + j];
                tmp[j] = a0 * bflo(vp.x) + a1 * bfhi(vp.x)
                       + a2 * bflo(vp.y) + a3 * bfhi(vp.y);
            }
            float4 res; res.x = tmp[0]; res.y = tmp[1]; res.z = tmp[2]; res.w = tmp[3];
            xo[d4] = res;
        }
    }
}

extern "C" void kernel_launch(void* const* d_in, const int* in_sizes, int n_in,
                              void* d_out, int out_size, void* d_ws, size_t ws_size,
                              hipStream_t stream) {
    const float* q  = (const float*)d_in[0];
    const float* k  = (const float*)d_in[1];
    const float* v  = (const float*)d_in[2];
    const float* ab = (const float*)d_in[3];
    const float* Wq = (const float*)d_in[4];
    const float* bq = (const float*)d_in[5];
    const float* Wk = (const float*)d_in[6];
    const float* bk = (const float*)d_in[7];
    const float* Wv = (const float*)d_in[8];
    const float* bv = (const float*)d_in[9];
    float* out = (float*)d_out;

    dim3 grid(B_TOTAL / MT), block(TPB);
    mhea_fused_kernel<<<grid, block, 0, stream>>>(q, k, v, ab, Wq, bq, Wk, bk, Wv, bv, out);
}
